// TextEncoder_9019431322230
// MI455X (gfx1250) — matrix operand order repaired and checked by reading, not compiled, not run on hardware
//
#include <hip/hip_runtime.h>

#define B_     32
#define T_     1024
#define C_     512
#define K_     5
#define DEPTH_ 3
#define H_     256
#define G_     1024      // 4*H
#define TP_    1028      // T + 4 (2 pad rows each side for K=5 conv)
#define EPS_   1e-5f
#define SLOPE_ 0.2f

typedef __attribute__((ext_vector_type(16))) __bf16 v16bf;
typedef __attribute__((ext_vector_type(8)))  float  v8f;

union FragBF { uint4 u[2]; v16bf v; };

__device__ __forceinline__ unsigned short f2bf(float f) {
    unsigned int u = __float_as_uint(f);
    u += 0x7fffu + ((u >> 16) & 1u);          // round-to-nearest-even
    return (unsigned short)(u >> 16);
}

__device__ __forceinline__ v8f v8zero() {
    v8f r;
#pragma unroll
    for (int i = 0; i < 8; ++i) r[i] = 0.0f;
    return r;
}

__device__ __forceinline__ v8f wmma_bf16(v16bf a, v16bf b, v8f c) {
    return __builtin_amdgcn_wmma_f32_16x16x32_bf16(false, a, false, b, (short)0, c,
                                                   false, false);
}

__device__ __forceinline__ float sigmoidf_(float x) {
    return 1.0f / (1.0f + __expf(-x));
}

// ---------------------------------------------------------------------------
// 1) conv weight transpose + bf16:  conv_w[d][k][ci][co] -> wt[d][k][co][ci]
// ---------------------------------------------------------------------------
__global__ void prep_convw_kernel(const float* __restrict__ cw,
                                  unsigned short* __restrict__ wt) {
    long idx = (long)blockIdx.x * blockDim.x + threadIdx.x;   // over 3*5*512*512
    if (idx >= (long)DEPTH_ * K_ * C_ * C_) return;
    int co = (int)(idx & 511);
    int ci = (int)((idx >> 9) & 511);
    int dk = (int)(idx >> 18);
    wt[((long)dk * C_ + co) * C_ + ci] = f2bf(cw[idx]);
}

// ---------------------------------------------------------------------------
// 2) LSTM combined weight transpose + bf16:
//    wcomb[dir][g][0..512)  = wx_dir[k][g],  wcomb[dir][g][512..768) = wh_dir[k-512][g]
// ---------------------------------------------------------------------------
__global__ void prep_wcomb_kernel(const float* __restrict__ wxf,
                                  const float* __restrict__ whf,
                                  const float* __restrict__ wxb,
                                  const float* __restrict__ whb,
                                  unsigned short* __restrict__ wc) {
    long idx = (long)blockIdx.x * blockDim.x + threadIdx.x;   // over 2*1024*768
    if (idx >= 2L * G_ * 768) return;
    int kk  = (int)(idx % 768);
    int g   = (int)((idx / 768) & 1023);
    int dir = (int)(idx / (768L * G_));
    const float* wx = dir ? wxb : wxf;
    const float* wh = dir ? whb : whf;
    float v = (kk < C_) ? wx[(long)kk * G_ + g] : wh[(long)(kk - C_) * G_ + g];
    wc[idx] = f2bf(v);
}

// ---------------------------------------------------------------------------
// 3) zero a bf16 buffer (pad rows must be zero; simplest: zero whole buffer)
// ---------------------------------------------------------------------------
__global__ void zero_bf16_kernel(unsigned short* __restrict__ p, long nhalf) {
    long i = ((long)blockIdx.x * blockDim.x + threadIdx.x) * 8;
    if (i + 8 <= nhalf) *(uint4*)(p + i) = make_uint4(0u, 0u, 0u, 0u);
}

// ---------------------------------------------------------------------------
// 4) embedding gather + mask -> padded bf16 activations [B][TP][C]
// ---------------------------------------------------------------------------
__global__ void embed_kernel(const int* __restrict__ x,
                             const unsigned char* __restrict__ m,
                             const float* __restrict__ emb,
                             unsigned short* __restrict__ pad) {
    int bt = blockIdx.x;                 // 0..B*T
    int b = bt >> 10, t = bt & 1023;
    int sym = x[bt];
    int mm = m[bt];
    const float* er = emb + (long)sym * C_;
    unsigned short* row = pad + ((long)b * TP_ + t + 2) * C_;
    int c = threadIdx.x;
    row[c]       = mm ? (unsigned short)0 : f2bf(er[c]);
    row[c + 256] = mm ? (unsigned short)0 : f2bf(er[c + 256]);
}

// ---------------------------------------------------------------------------
// 5) conv1d as implicit GEMM via WMMA bf16.
//    One wave per 16(co) x 16(t) output tile of one batch.  K = 5 * 512.
//    raw[b][t][co] (f32) = sum_k sum_ci wt[d][k][co][ci] * inpad[b][t+k][ci] + bias
// ---------------------------------------------------------------------------
__global__ __launch_bounds__(32)
void conv_gemm_kernel(const unsigned short* __restrict__ inpad,
                      const unsigned short* __restrict__ wt,
                      const float* __restrict__ cb,
                      float* __restrict__ raw, int d) {
    int lane   = threadIdx.x;
    int lane16 = lane & 15;
    int hi     = lane >> 4;
    int co_base = blockIdx.x << 4;
    int t_base  = blockIdx.y << 4;
    int b       = blockIdx.z;

    v8f acc = v8zero();
#pragma unroll
    for (int k = 0; k < K_; ++k) {
        // B fragment source: lane's column n = time position (t_base+lane16),
        // ISA layout chunks at ci = kc*32 + 8*hi and kc*32 + 16 + 8*hi
        const unsigned short* prow =
            inpad + ((long)b * TP_ + t_base + lane16 + k) * C_ + 8 * hi;
        // A fragment source: lane's row m = co, ISA A-layout chunks at
        // ci = kc*32 + 8*hi and kc*32 + 16 + 8*hi
        const unsigned short* wrow =
            wt + (((long)(d * K_ + k) * C_) + co_base + lane16) * C_ + 8 * hi;
#pragma unroll
        for (int cc = 0; cc < 16; ++cc) {
            FragBF bfr, af;
            bfr.u[0] = *(const uint4*)(prow + cc * 32);
            bfr.u[1] = *(const uint4*)(prow + cc * 32 + 16);
            af.u[0]  = *(const uint4*)(wrow + cc * 32);
            af.u[1]  = *(const uint4*)(wrow + cc * 32 + 16);
            acc = wmma_bf16(af.v, bfr.v, acc);
        }
    }
    // D layout: lane holds n = lane16 (time), m = v + 8*hi (co) -> contiguous co
    int co = co_base + 8 * hi;
    const float* bp = cb + d * C_ + co;
    float* o = raw + ((long)b * T_ + t_base + lane16) * C_ + co;
    float4 o0 = make_float4(acc[0] + bp[0], acc[1] + bp[1], acc[2] + bp[2], acc[3] + bp[3]);
    float4 o1 = make_float4(acc[4] + bp[4], acc[5] + bp[5], acc[6] + bp[6], acc[7] + bp[7]);
    *(float4*)(o)     = o0;
    *(float4*)(o + 4) = o1;
}

// ---------------------------------------------------------------------------
// 6) LayerNorm over C + leaky relu + mask -> padded bf16 for next layer
// ---------------------------------------------------------------------------
__global__ __launch_bounds__(256)
void norm_kernel(const float* __restrict__ raw,
                 const unsigned char* __restrict__ m,
                 const float* __restrict__ gamma,
                 const float* __restrict__ beta,
                 unsigned short* __restrict__ padout, int d) {
    __shared__ float s1[256], s2[256];
    int bt = blockIdx.x;
    int b = bt >> 10, t = bt & 1023;
    int tid = threadIdx.x;
    const float* r = raw + (long)bt * C_;
    float v0 = r[tid], v1 = r[tid + 256];
    s1[tid] = v0 + v1;
    s2[tid] = v0 * v0 + v1 * v1;
    __syncthreads();
#pragma unroll
    for (int s = 128; s > 0; s >>= 1) {
        if (tid < s) { s1[tid] += s1[tid + s]; s2[tid] += s2[tid + s]; }
        __syncthreads();
    }
    float mu   = s1[0] * (1.0f / C_);
    float var  = s2[0] * (1.0f / C_) - mu * mu;
    float rstd = rsqrtf(var + EPS_);
    int mm = m[bt];
    unsigned short* o = padout + ((long)b * TP_ + t + 2) * C_;
    const float* gg = gamma + d * C_;
    const float* bb = beta + d * C_;
#pragma unroll
    for (int j = 0; j < 2; ++j) {
        int c = tid + j * 256;
        float v = (j == 0) ? v0 : v1;
        float y = (v - mu) * rstd * gg[c] + bb[c];
        y = (y > 0.0f) ? y : SLOPE_ * y;
        if (mm) y = 0.0f;
        o[c] = f2bf(y);
    }
}

// ---------------------------------------------------------------------------
// 7) Bidirectional LSTM: one 1024-thread workgroup per direction.
//    Per step: z = x_t @ Wx + h_{t-1} @ Wh + bias computed fully in WMMA
//    (combined K = 512 + 256), h state double-buffered bf16 in LDS.
//    Wave w owns the (16 hh x 16 b) block for all 4 gates.
// ---------------------------------------------------------------------------
__global__ __launch_bounds__(1024)
void lstm_kernel(const unsigned short* __restrict__ act,   // padded [B][TP][C] bf16
                 const unsigned short* __restrict__ wc,    // [2][G][768] bf16
                 const float* __restrict__ bf_,
                 const float* __restrict__ bb_,
                 float* __restrict__ hf,                    // [T][B][H] f32
                 float* __restrict__ hb) {
    __shared__ __align__(16) unsigned short hl[2][B_][H_];

    int dir  = blockIdx.x;
    int tid  = threadIdx.x;
    int wave = tid >> 5, lane = tid & 31;
    int lane16 = lane & 15, hi = lane >> 4;
    int hh_base = (wave & 15) << 4;
    int b_base  = (wave >> 4) << 4;
    int b = b_base + lane16;

    const float* bias = dir ? bb_ : bf_;
    float* hout = dir ? hb : hf;

    // zero h_{-1} (both LDS buffers): 16384 halves = 2048 uint4
    {
        uint4* p = (uint4*)&hl[0][0][0];
        uint4 z = make_uint4(0u, 0u, 0u, 0u);
        p[tid] = z;
        p[tid + 1024] = z;
    }

    // bias vector per gate tile (constant across steps); element v -> g = q*H + hh_base + 8*hi + v
    v8f biasv[4];
#pragma unroll
    for (int q = 0; q < 4; ++q) {
        const float* bp = bias + q * H_ + hh_base + 8 * hi;
#pragma unroll
        for (int v = 0; v < 8; ++v) biasv[q][v] = bp[v];
    }

    // A-matrix row base per gate tile: row g = q*H + hh_base + lane16
    const unsigned short* arow[4];
#pragma unroll
    for (int q = 0; q < 4; ++q)
        arow[q] = wc + (long)(dir * G_ + q * H_ + hh_base + lane16) * 768;

    v8f cst = v8zero();          // cell state block (f32, in registers)
    __syncthreads();

    int cur = 0;
    int t  = dir ? (T_ - 1) : 0;
    int dt = dir ? -1 : 1;

    for (int s = 0; s < T_; ++s, t += dt) {
        v8f acc[4];
#pragma unroll
        for (int q = 0; q < 4; ++q) acc[q] = v8zero();

        // --- input projection: K chunks 0..15 over C=512 from global bf16 acts
        const unsigned short* pb = act + ((long)b * TP_ + t + 2) * C_ + 8 * hi;
#pragma unroll 4
        for (int kc = 0; kc < 16; ++kc) {
            FragBF bfr;
            bfr.u[0] = *(const uint4*)(pb + kc * 32);
            bfr.u[1] = *(const uint4*)(pb + kc * 32 + 16);
#pragma unroll
            for (int q = 0; q < 4; ++q) {
                FragBF af;
                const unsigned short* ap = arow[q] + kc * 32 + 8 * hi;
                af.u[0] = *(const uint4*)(ap);
                af.u[1] = *(const uint4*)(ap + 16);
                acc[q] = wmma_bf16(af.v, bfr.v, acc[q]);
            }
        }
        // --- recurrent projection: K chunks over H=256 from LDS h_{t-1}
        const unsigned short* ph = &hl[cur][b][8 * hi];
#pragma unroll
        for (int kc = 0; kc < 8; ++kc) {
            FragBF bfr;
            bfr.u[0] = *(const uint4*)(ph + kc * 32);
            bfr.u[1] = *(const uint4*)(ph + kc * 32 + 16);
#pragma unroll
            for (int q = 0; q < 4; ++q) {
                FragBF af;
                const unsigned short* ap = arow[q] + C_ + kc * 32 + 8 * hi;
                af.u[0] = *(const uint4*)(ap);
                af.u[1] = *(const uint4*)(ap + 16);
                acc[q] = wmma_bf16(af.v, bfr.v, acc[q]);
            }
        }

        // --- gates + state update (lane owns b fixed, hh = hh_base + 8*hi + v)
        v8f hnew;
#pragma unroll
        for (int v = 0; v < 8; ++v) {
            float zi = acc[0][v] + biasv[0][v];
            float zf = acc[1][v] + biasv[1][v];
            float zg = acc[2][v] + biasv[2][v];
            float zo = acc[3][v] + biasv[3][v];
            float ig = sigmoidf_(zi);
            float fg = sigmoidf_(zf);
            float gg = tanhf(zg);
            float og = sigmoidf_(zo);
            float cc = fg * cst[v] + ig * gg;
            cst[v] = cc;
            hnew[v] = og * tanhf(cc);
        }

        // h -> global f32 [t][b][hh] (contiguous 8 floats per lane)
        float* ho = hout + ((long)t * B_ + b) * H_ + hh_base + 8 * hi;
        *(float4*)(ho)     = make_float4(hnew[0], hnew[1], hnew[2], hnew[3]);
        *(float4*)(ho + 4) = make_float4(hnew[4], hnew[5], hnew[6], hnew[7]);

        // h -> LDS bf16 double buffer for next step
        uint4 pk;
        pk.x = (unsigned)f2bf(hnew[0]) | ((unsigned)f2bf(hnew[1]) << 16);
        pk.y = (unsigned)f2bf(hnew[2]) | ((unsigned)f2bf(hnew[3]) << 16);
        pk.z = (unsigned)f2bf(hnew[4]) | ((unsigned)f2bf(hnew[5]) << 16);
        pk.w = (unsigned)f2bf(hnew[6]) | ((unsigned)f2bf(hnew[7]) << 16);
        *(uint4*)&hl[cur ^ 1][b][hh_base + 8 * hi] = pk;

        __syncthreads();
        cur ^= 1;
    }
}

// ---------------------------------------------------------------------------
// 8) concat hf/hb, transpose to (B, 2H, T), mask
// ---------------------------------------------------------------------------
__global__ __launch_bounds__(256)
void final_kernel(const float* __restrict__ hf,
                  const float* __restrict__ hb,
                  const unsigned char* __restrict__ m,
                  float* __restrict__ out) {
    long idx = (long)blockIdx.x * blockDim.x + threadIdx.x;   // over B*512*T
    int t = (int)(idx & 1023);
    int c = (int)((idx >> 10) & 511);
    int b = (int)(idx >> 19);
    float v = (c < H_) ? hf[((long)t * B_ + b) * H_ + c]
                       : hb[((long)t * B_ + b) * H_ + (c - H_)];
    if (m[b * T_ + t]) v = 0.0f;
    out[idx] = v;
}

// ---------------------------------------------------------------------------
extern "C" void kernel_launch(void* const* d_in, const int* in_sizes, int n_in,
                              void* d_out, int out_size, void* d_ws, size_t ws_size,
                              hipStream_t stream) {
    const int*           x     = (const int*)d_in[0];
    const unsigned char* m     = (const unsigned char*)d_in[2];
    const float*         emb   = (const float*)d_in[3];
    const float*         cw    = (const float*)d_in[4];
    const float*         cbias = (const float*)d_in[5];
    const float*         gamma = (const float*)d_in[6];
    const float*         beta  = (const float*)d_in[7];
    const float*         wxf   = (const float*)d_in[8];
    const float*         whf   = (const float*)d_in[9];
    const float*         bfv   = (const float*)d_in[10];
    const float*         wxb   = (const float*)d_in[11];
    const float*         whb   = (const float*)d_in[12];
    const float*         bbv   = (const float*)d_in[13];
    float*               out   = (float*)d_out;

    char* w = (char*)d_ws;
    size_t off = 0;
    auto take = [&](size_t bytes) -> void* {
        void* p = (void*)(w + off);
        off = (off + bytes + 255) & ~(size_t)255;
        return p;
    };
    const long NWTC  = (long)DEPTH_ * K_ * C_ * C_;   // 3,932,160 halves
    const long NWC   = 2L * G_ * 768;                 // 1,572,864 halves
    const long NPAD  = (long)B_ * TP_ * C_;           // 16,842,752 halves
    const long NRAW  = (long)B_ * T_ * C_;            // 16,777,216 floats
    const long NH    = (long)T_ * B_ * H_;            // 8,388,608 floats

    unsigned short* wtc   = (unsigned short*)take(NWTC * 2);
    unsigned short* wcomb = (unsigned short*)take(NWC * 2);
    unsigned short* padA  = (unsigned short*)take(NPAD * 2);
    unsigned short* padB  = (unsigned short*)take(NPAD * 2);
    float*          raw   = (float*)take(NRAW * 4);
    float*          hfbuf = (float*)take(NH * 4);
    float*          hbbuf = (float*)take(NH * 4);

    // weight prep (bf16 transposes)
    prep_convw_kernel<<<(int)((NWTC + 255) / 256), 256, 0, stream>>>(cw, wtc);
    prep_wcomb_kernel<<<(int)((NWC + 255) / 256), 256, 0, stream>>>(wxf, whf, wxb, whb, wcomb);

    // zero padded activation buffers (pad rows must be 0)
    zero_bf16_kernel<<<(int)((NPAD / 8 + 255) / 256), 256, 0, stream>>>(padA, NPAD);
    zero_bf16_kernel<<<(int)((NPAD / 8 + 255) / 256), 256, 0, stream>>>(padB, NPAD);

    // embedding + mask
    embed_kernel<<<B_ * T_, 256, 0, stream>>>(x, m, emb, padA);

    // conv stack (ping-pong padA <-> padB)
    dim3 cgrid(C_ / 16, T_ / 16, B_);
    conv_gemm_kernel<<<cgrid, 32, 0, stream>>>(padA, wtc, cbias, raw, 0);
    norm_kernel<<<B_ * T_, 256, 0, stream>>>(raw, m, gamma, beta, padB, 0);
    conv_gemm_kernel<<<cgrid, 32, 0, stream>>>(padB, wtc, cbias, raw, 1);
    norm_kernel<<<B_ * T_, 256, 0, stream>>>(raw, m, gamma, beta, padA, 1);
    conv_gemm_kernel<<<cgrid, 32, 0, stream>>>(padA, wtc, cbias, raw, 2);
    norm_kernel<<<B_ * T_, 256, 0, stream>>>(raw, m, gamma, beta, padB, 2);

    // bidirectional LSTM: block 0 = forward, block 1 = backward
    lstm_kernel<<<2, 1024, 0, stream>>>(padB, wcomb, bfv, bbv, hfbuf, hbbuf);

    // concat + transpose + mask
    final_kernel<<<(int)((B_ * 512L * T_) / 256), 256, 0, stream>>>(hfbuf, hbbuf, m, out);
}
